// DeepSetHyperRNNAgent_68513318306042
// MI455X (gfx1250) — compile-verified, matrix operand order checked by reading, not run on hardware
//
#include <hip/hip_runtime.h>

// ---------------------------------------------------------------------------
// DeepSetHyperRNNAgent fused kernel for gfx1250 (wave32, WMMA bf16).
// One wave processes a 16-row tile. All GEMMs via v_wmma_f32_16x16x32_bf16.
// Weights are prepacked once per call into d_ws as zero-padded bf16 in
// (N,K) fragment-friendly layout, so each B fragment is two b128 loads.
// ---------------------------------------------------------------------------

typedef __attribute__((ext_vector_type(16))) __bf16 v16bf;
typedef __attribute__((ext_vector_type(8)))  float  v8f;

#define WAVES_PER_BLK 4

// bf16 element offsets inside d_ws
#define OFF_WCAT 0                       // 64 x 96  : [W1_own | W1_enemy | W1_ally | pad]
#define OFF_WIH  (OFF_WCAT + 64 * 96)    // 192 x 64
#define OFF_WHH  (OFF_WIH + 192 * 64)    // 192 x 64
#define OFF_W2   (OFF_WHH + 192 * 64)    // 16 x 64  (rows 6..15 zero)
#define OFF_WH1  (OFF_W2 + 16 * 64)      // 64 x 32  (k 16..31 zero)
#define OFF_WG   (OFF_WH1 + 64 * 32)     // 64 x 64  = Wh2[:64,:]^T  (N=k-dim, K=h-dim)
#define WS_ELEMS (OFF_WG + 64 * 64)      // 37888 bf16 = 75776 bytes

__device__ inline v8f zero8() {
  v8f z;
#pragma unroll
  for (int i = 0; i < 8; ++i) z[i] = 0.f;
  return z;
}

__device__ inline v8f wmma_bf16(v16bf a, v16bf b, v8f c) {
  // (neg_a, A, neg_b, B, c_mod, C, reuse_a, reuse_b)
  return __builtin_amdgcn_wmma_f32_16x16x32_bf16(false, a, false, b, (short)0, c, false, false);
}

__device__ inline float sigmoidf(float x) { return 1.f / (1.f + __expf(-x)); }

// K index inside a 32-wide K step for 16-bit A/B fragments (ISA 7.12.2).
__device__ inline int frag_k(int i, int half) {
  return ((i < 4) ? 0 : 16) + half * 8 + 2 * (i & 3);
}

// A fragment: 16 rows from src (row-major fp32, stride ld), K window [k0,k0+32), pad k>=kvalid.
__device__ inline v16bf load_a(const float* src, int ld, int k0, int kvalid, int lane) {
  const int row = lane & 15, half = lane >> 4;
  v16bf a;
#pragma unroll
  for (int i = 0; i < 8; ++i) {
    int k = k0 + frag_k(i, half);
    float v0 = (k     < kvalid) ? src[row * ld + k]     : 0.f;
    float v1 = (k + 1 < kvalid) ? src[row * ld + k + 1] : 0.f;
    a[2 * i] = (__bf16)v0;
    a[2 * i + 1] = (__bf16)v1;
  }
  return a;
}

// B fragment from prepacked bf16 weight stored (N,K) row-major (K multiple of 32).
// Fragment element order makes each half a contiguous 8-element (16B) run.
__device__ inline v16bf load_b_pk(const __bf16* W, int ldw, int n0, int k0, int lane) {
  const int nl = lane & 15, half = lane >> 4;
  const __bf16* p = W + (size_t)(n0 + nl) * ldw + k0 + half * 8;
  v16bf b;
#pragma unroll
  for (int i = 0; i < 8; ++i) b[i] = p[i];         // k0 + half*8 + [0..7]
#pragma unroll
  for (int i = 0; i < 8; ++i) b[8 + i] = p[16 + i]; // k0 + 16 + half*8 + [0..7]
  return b;
}

// ---------------------------------------------------------------------------
// Prepack kernel: fp32 weights -> zero-padded bf16 (N,K) layouts in d_ws.
// ---------------------------------------------------------------------------
__global__ void pack_weights_kernel(
    const float* __restrict__ W1_own, const float* __restrict__ W1_enemy,
    const float* __restrict__ W1_ally,
    const float* __restrict__ W_ih, const float* __restrict__ W_hh,
    const float* __restrict__ W2, const float* __restrict__ Wh1,
    const float* __restrict__ Wh2, __bf16* __restrict__ ws) {
  const int t = blockIdx.x * blockDim.x + threadIdx.x;
  const int stride = gridDim.x * blockDim.x;
  for (int i = t; i < 64 * 96; i += stride) {  // Wcat
    int n = i / 96, k = i - n * 96;
    float v = 0.f;
    if (k < 48)      v = W1_own[n * 48 + k];
    else if (k < 64) v = W1_enemy[n * 16 + (k - 48)];
    else if (k < 80) v = W1_ally[n * 16 + (k - 64)];
    ws[OFF_WCAT + i] = (__bf16)v;
  }
  for (int i = t; i < 192 * 64; i += stride) {  // GRU weights (already (N,K))
    ws[OFF_WIH + i] = (__bf16)W_ih[i];
    ws[OFF_WHH + i] = (__bf16)W_hh[i];
  }
  for (int i = t; i < 16 * 64; i += stride) {  // W2 padded to 16 rows
    int n = i >> 6;
    ws[OFF_W2 + i] = (__bf16)(n < 6 ? W2[i] : 0.f);
  }
  for (int i = t; i < 64 * 32; i += stride) {  // Wh1 padded K 16->32
    int n = i >> 5, k = i & 31;
    ws[OFF_WH1 + i] = (__bf16)(k < 16 ? Wh1[n * 16 + k] : 0.f);
  }
  for (int i = t; i < 64 * 64; i += stride) {  // Wg = Wh2[:64,:]^T
    int n = i >> 6, c = i & 63;
    ws[OFF_WG + i] = (__bf16)Wh2[c * 64 + n];
  }
}

// ---------------------------------------------------------------------------
// Main fused kernel.
// ---------------------------------------------------------------------------
__global__ __launch_bounds__(WAVES_PER_BLK * 32) void agent_fused_kernel(
    const float* __restrict__ own, const float* __restrict__ enemy, const float* __restrict__ ally,
    const int* __restrict__ agent_idx, const int* __restrict__ act_idx,
    const float* __restrict__ hidden,
    const float* __restrict__ b1_own,
    const float* __restrict__ emb_agent, const float* __restrict__ emb_action,
    const float* __restrict__ b_ih, const float* __restrict__ b_hh,
    const float* __restrict__ b2,
    const float* __restrict__ bh1,
    const float* __restrict__ Wh2, const float* __restrict__ bh2,
    const __bf16* __restrict__ ws,
    float* __restrict__ out, int Btot) {
  __shared__ float bufX[WAVES_PER_BLK][16 * 80];  // Xin -> x -> h'; col 64 later holds c[r]
  __shared__ float bufH[WAVES_PER_BLK][16 * 64];  // hprev -> g

  const int lane = threadIdx.x & 31;
  const int wv = threadIdx.x >> 5;
  const int tile = blockIdx.x * WAVES_PER_BLK + wv;
  const int r0 = tile * 16;
  if (r0 >= Btot) return;

  float* Xs = bufX[wv];
  float* Hs = bufH[wv];
  const int nlane = lane & 15;
  const int halfsel = lane >> 4;

  // ---- Phase 1: stage augmented input Xin = [own(48) | sum_e enemy(16) | sum_a ally(16)]
#pragma unroll
  for (int i = 0; i < 24; ++i) {
    int o = lane + 32 * i;  // 768 own elements
    int row = o / 48, col = o - row * 48;
    Xs[row * 80 + col] = own[(size_t)(r0 + row) * 48 + col];
  }
#pragma unroll
  for (int i = 0; i < 8; ++i) {
    int o = lane + 32 * i;  // 256 (row,d) reductions
    int row = o >> 4, d = o & 15;
    const float* ep = enemy + (size_t)(r0 + row) * 256 + d;
    float se = 0.f;
#pragma unroll
    for (int e = 0; e < 16; ++e) se += ep[e * 16];
    Xs[row * 80 + 48 + d] = se;
    const float* ap = ally + (size_t)(r0 + row) * 240 + d;
    float sa = 0.f;
#pragma unroll
    for (int e = 0; e < 15; ++e) sa += ap[e * 16];
    Xs[row * 80 + 64 + d] = sa;
  }

  // ---- Phase 2: x = relu(Xin @ Wcat^T + b1_own + emb_agent[idx] + emb_action[idx])
  v16bf aemb[3];
#pragma unroll
  for (int ks = 0; ks < 3; ++ks) aemb[ks] = load_a(Xs, 80, ks * 32, 80, lane);

  v8f xf[4];
#pragma unroll
  for (int nt = 0; nt < 4; ++nt) {
    v8f acc = zero8();
#pragma unroll
    for (int ks = 0; ks < 3; ++ks)
      acc = wmma_bf16(aemb[ks], load_b_pk(ws + OFF_WCAT, 96, nt * 16, ks * 32, lane), acc);
    const int col = nt * 16 + nlane;
    const float bo = b1_own[col];
#pragma unroll
    for (int j = 0; j < 8; ++j) {
      int row = r0 + j + halfsel * 8;
      float v = acc[j] + bo + emb_agent[(size_t)agent_idx[row] * 64 + col] +
                emb_action[(size_t)act_idx[row] * 64 + col];
      acc[j] = v > 0.f ? v : 0.f;
    }
    xf[nt] = acc;
  }
#pragma unroll
  for (int nt = 0; nt < 4; ++nt) {
    const int col = nt * 16 + nlane;
#pragma unroll
    for (int j = 0; j < 8; ++j) Xs[(j + halfsel * 8) * 80 + col] = xf[nt][j];
  }
  // stage hprev (contiguous 1024 floats)
#pragma unroll
  for (int i = 0; i < 32; ++i) {
    int o = lane + 32 * i;
    Hs[o] = hidden[(size_t)r0 * 64 + o];
  }

  // ---- Phase 3: GRU cell
  v16bf xa[2], ha[2];
#pragma unroll
  for (int ks = 0; ks < 2; ++ks) {
    xa[ks] = load_a(Xs, 80, ks * 32, 64, lane);
    ha[ks] = load_a(Hs, 64, ks * 32, 64, lane);
  }
  v8f hnew[4];
#pragma unroll
  for (int nt = 0; nt < 4; ++nt) {
    v8f gir = zero8(), giz = zero8(), gin = zero8();
    v8f ghr = zero8(), ghz = zero8(), ghn = zero8();
#pragma unroll
    for (int ks = 0; ks < 2; ++ks) {
      gir = wmma_bf16(xa[ks], load_b_pk(ws + OFF_WIH, 64, nt * 16,       ks * 32, lane), gir);
      giz = wmma_bf16(xa[ks], load_b_pk(ws + OFF_WIH, 64, 64 + nt * 16,  ks * 32, lane), giz);
      gin = wmma_bf16(xa[ks], load_b_pk(ws + OFF_WIH, 64, 128 + nt * 16, ks * 32, lane), gin);
      ghr = wmma_bf16(ha[ks], load_b_pk(ws + OFF_WHH, 64, nt * 16,       ks * 32, lane), ghr);
      ghz = wmma_bf16(ha[ks], load_b_pk(ws + OFF_WHH, 64, 64 + nt * 16,  ks * 32, lane), ghz);
      ghn = wmma_bf16(ha[ks], load_b_pk(ws + OFF_WHH, 64, 128 + nt * 16, ks * 32, lane), ghn);
    }
    const int col = nt * 16 + nlane;
    const float bir = b_ih[col], biz = b_ih[64 + col], bin = b_ih[128 + col];
    const float bhr = b_hh[col], bhz = b_hh[64 + col], bhn = b_hh[128 + col];
    v8f hn;
#pragma unroll
    for (int j = 0; j < 8; ++j) {
      int m = j + halfsel * 8;
      float rg = sigmoidf((gir[j] + bir) + (ghr[j] + bhr));
      float zg = sigmoidf((giz[j] + biz) + (ghz[j] + bhz));
      float ng = tanhf((gin[j] + bin) + rg * (ghn[j] + bhn));
      float hp = Hs[m * 64 + col];
      hn[j] = (1.f - zg) * ng + zg * hp;
    }
    hnew[nt] = hn;
  }
  // store h' into Xs and write h output (contiguous)
#pragma unroll
  for (int nt = 0; nt < 4; ++nt) {
    const int col = nt * 16 + nlane;
#pragma unroll
    for (int j = 0; j < 8; ++j) Xs[(j + halfsel * 8) * 80 + col] = hnew[nt][j];
  }
  const size_t qbase = (size_t)Btot * 22;
#pragma unroll
  for (int i = 0; i < 32; ++i) {
    int o = lane + 32 * i;
    out[qbase + (size_t)r0 * 64 + o] = Xs[(o >> 6) * 80 + (o & 63)];
  }

  // c[r] = h'[r] . bh2[:64] + bh2[64]
  if (lane < 16) {
    float c = bh2[64];
#pragma unroll
    for (int h = 0; h < 64; ++h) c += Xs[lane * 80 + h] * bh2[h];
    Xs[lane * 80 + 64] = c;
  }

  // ---- Phase 4: q_normal = h'@W2^T + b2 ; g = h'@Wh2[:64] + Wh2[64]
  v16bf hpa[2];
#pragma unroll
  for (int ks = 0; ks < 2; ++ks) hpa[ks] = load_a(Xs, 80, ks * 32, 64, lane);
  {
    v8f acc = zero8();
#pragma unroll
    for (int ks = 0; ks < 2; ++ks)
      acc = wmma_bf16(hpa[ks], load_b_pk(ws + OFF_W2, 64, 0, ks * 32, lane), acc);
    if (nlane < 6) {
      const float bb = b2[nlane];
#pragma unroll
      for (int j = 0; j < 8; ++j)
        out[(size_t)(r0 + j + halfsel * 8) * 22 + nlane] = acc[j] + bb;
    }
  }
#pragma unroll
  for (int nt = 0; nt < 4; ++nt) {
    v8f acc = zero8();
#pragma unroll
    for (int ks = 0; ks < 2; ++ks)
      acc = wmma_bf16(hpa[ks], load_b_pk(ws + OFF_WG, 64, nt * 16, ks * 32, lane), acc);
    const int col = nt * 16 + nlane;
    const float wlast = Wh2[64 * 64 + col];
#pragma unroll
    for (int j = 0; j < 8; ++j) Hs[(j + halfsel * 8) * 64 + col] = acc[j] + wlast;
  }

  // ---- Phase 5: per-row hypernet + attack Q:
  //   q_attack[r][e] = sum_k relu(enemy[r]@Wh1^T + bh1)[e][k] * g[r][k] + c[r]
  v16bf bhyp[4];
  float bh1c[4];
#pragma unroll
  for (int nt = 0; nt < 4; ++nt) {
    bhyp[nt] = load_b_pk(ws + OFF_WH1, 32, nt * 16, 0, lane);
    bh1c[nt] = bh1[nt * 16 + nlane];
  }
  for (int r = 0; r < 16; ++r) {
    v16bf ae = load_a(enemy + (size_t)(r0 + r) * 256, 16, 0, 16, lane);
    float part[8];
#pragma unroll
    for (int j = 0; j < 8; ++j) part[j] = 0.f;
#pragma unroll
    for (int nt = 0; nt < 4; ++nt) {
      v8f acc = zero8();
      acc = wmma_bf16(ae, bhyp[nt], acc);
      const float gv = Hs[r * 64 + nt * 16 + nlane];
#pragma unroll
      for (int j = 0; j < 8; ++j) {
        float v = acc[j] + bh1c[nt];
        v = v > 0.f ? v : 0.f;
        part[j] += v * gv;
      }
    }
#pragma unroll
    for (int j = 0; j < 8; ++j) {
      part[j] += __shfl_xor(part[j], 1, 32);
      part[j] += __shfl_xor(part[j], 2, 32);
      part[j] += __shfl_xor(part[j], 4, 32);
      part[j] += __shfl_xor(part[j], 8, 32);
    }
    const float cr = Xs[r * 80 + 64];
    if (nlane == 0) {
      const int e0 = halfsel * 8;
#pragma unroll
      for (int j = 0; j < 8; ++j)
        out[(size_t)(r0 + r) * 22 + 6 + e0 + j] = part[j] + cr;
    }
  }
}

extern "C" void kernel_launch(void* const* d_in, const int* in_sizes, int n_in,
                              void* d_out, int out_size, void* d_ws, size_t ws_size,
                              hipStream_t stream) {
  const float* own        = (const float*)d_in[1];
  const float* enemy      = (const float*)d_in[2];
  const float* ally       = (const float*)d_in[3];
  const int*   aidx       = (const int*)d_in[4];
  const int*   lidx       = (const int*)d_in[5];
  const float* hidden     = (const float*)d_in[6];
  const float* W1_own     = (const float*)d_in[7];
  const float* b1_own     = (const float*)d_in[8];
  const float* W1_enemy   = (const float*)d_in[9];
  const float* W1_ally    = (const float*)d_in[10];
  const float* emb_agent  = (const float*)d_in[11];
  const float* emb_action = (const float*)d_in[12];
  const float* W_ih       = (const float*)d_in[13];
  const float* W_hh       = (const float*)d_in[14];
  const float* b_ih       = (const float*)d_in[15];
  const float* b_hh       = (const float*)d_in[16];
  const float* W2         = (const float*)d_in[17];
  const float* b2         = (const float*)d_in[18];
  const float* Wh1        = (const float*)d_in[19];
  const float* bh1        = (const float*)d_in[20];
  const float* Wh2        = (const float*)d_in[21];
  const float* bh2        = (const float*)d_in[22];
  float* out = (float*)d_out;
  __bf16* ws = (__bf16*)d_ws;

  const int Btot = in_sizes[6] / 64;  // hidden_state is (B, 64)
  const int tiles = (Btot + 15) / 16;
  const int blocks = (tiles + WAVES_PER_BLK - 1) / WAVES_PER_BLK;

  // Prepack weights to bf16 (N,K) layouts in workspace (~76 KB).
  hipLaunchKernelGGL(pack_weights_kernel, dim3(64), dim3(256), 0, stream,
                     W1_own, W1_enemy, W1_ally, W_ih, W_hh, W2, Wh1, Wh2, ws);

  hipLaunchKernelGGL(agent_fused_kernel, dim3(blocks), dim3(WAVES_PER_BLK * 32), 0, stream,
                     own, enemy, ally, aidx, lidx, hidden,
                     b1_own, emb_agent, emb_action,
                     b_ih, b_hh, b2, bh1, Wh2, bh2, ws,
                     out, Btot);
}
